// SparseMemoryAttention_64055142252572
// MI455X (gfx1250) — matrix operand order repaired and checked by reading, not compiled, not run on hardware
//
#include <hip/hip_runtime.h>

#define DEVI __device__ __forceinline__

typedef unsigned short u16;
typedef __attribute__((ext_vector_type(16))) __bf16 v16bf;
typedef __attribute__((ext_vector_type(8)))  float  v8f;

namespace {
constexpr int kB = 2, kS = 2048, kE = 1024, kH = 16, kD = 64, kHalf = 128;
constexpr int kM = kB * kS;  // 4096 rows
}

union Frag {
  int4 i4[2];
  u16 u[16];
  v16bf v;
};

DEVI v8f vzero() {
  v8f z;
#pragma unroll
  for (int i = 0; i < 8; ++i) z[i] = 0.f;
  return z;
}

// Split fp32 into bf16 hi (truncation -> remainder exact) + bf16 lo.
DEVI void split1(float x, u16& h, u16& l) {
  unsigned u = __float_as_uint(x);
  h = (u16)(u >> 16);
  float r = x - __uint_as_float(u & 0xffff0000u);
  l = (u16)(__float_as_uint(r) >> 16);
}

DEVI void split16(const float* x, v16bf& hi, v16bf& lo) {
  Frag fh, fl;
#pragma unroll
  for (int i = 0; i < 16; ++i) split1(x[i], fh.u[i], fl.u[i]);
  hi = fh.v;
  lo = fl.v;
}

DEVI v8f wmma_bf16(v16bf a, v16bf b, v8f c) {
  return __builtin_amdgcn_wmma_f32_16x16x32_bf16(false, a, false, b, (short)0, c,
                                                 false, false);
}

// CDNA5 async global->LDS copy (16B per lane), tracked by ASYNCcnt.
DEVI void async16(unsigned lds_off, const u16* base, unsigned byte_off) {
  asm volatile("global_load_async_to_lds_b128 %0, %1, %2"
               :
               : "v"(lds_off), "v"(byte_off), "s"(base)
               : "memory");
}
DEVI void wait_async0() { asm volatile("s_wait_asynccnt 0" ::: "memory"); }
DEVI void wait_ds0() { asm volatile("s_wait_dscnt 0" ::: "memory"); }

// ---------------------------------------------------------------------------
// Split fp32 array into bf16 hi/lo planes (same layout), float4-vectorized.
// ---------------------------------------------------------------------------
__global__ __launch_bounds__(256) void split_planes(const float* __restrict__ in,
                                                    u16* __restrict__ hi,
                                                    u16* __restrict__ lo, int n4) {
  int i = blockIdx.x * blockDim.x + threadIdx.x;
  if (i >= n4) return;
  float4 f = ((const float4*)in)[i];
  u16 h0, h1, h2, h3, l0, l1, l2, l3;
  split1(f.x, h0, l0);
  split1(f.y, h1, l1);
  split1(f.z, h2, l2);
  split1(f.w, h3, l3);
  ((uint2*)hi)[i] = make_uint2((unsigned)h0 | ((unsigned)h1 << 16),
                               (unsigned)h2 | ((unsigned)h3 << 16));
  ((uint2*)lo)[i] = make_uint2((unsigned)l0 | ((unsigned)l1 << 16),
                               (unsigned)l2 | ((unsigned)l3 << 16));
}

// ---------------------------------------------------------------------------
// Split + transpose W[1024][1024] -> Wt hi/lo planes [N][K] (bf16).
// ---------------------------------------------------------------------------
__global__ __launch_bounds__(256) void split_transpose(const float* __restrict__ W,
                                                       u16* __restrict__ WtHi,
                                                       u16* __restrict__ WtLo) {
  __shared__ __align__(16) u16 sHi[64][64];
  __shared__ __align__(16) u16 sLo[64][64];
  const int tid = threadIdx.x;
  const int n0 = blockIdx.x * 64, k0 = blockIdx.y * 64;
#pragma unroll
  for (int i = 0; i < 4; ++i) {
    int idx = tid + 256 * i;
    int r = idx >> 4, c4 = (idx & 15) * 4;  // r = k-local, c4 = n-local
    const float4 f = *(const float4*)(W + (size_t)(k0 + r) * kE + n0 + c4);
    u16 h, l;
    split1(f.x, h, l); sHi[c4 + 0][r] = h; sLo[c4 + 0][r] = l;
    split1(f.y, h, l); sHi[c4 + 1][r] = h; sLo[c4 + 1][r] = l;
    split1(f.z, h, l); sHi[c4 + 2][r] = h; sLo[c4 + 2][r] = l;
    split1(f.w, h, l); sHi[c4 + 3][r] = h; sLo[c4 + 3][r] = l;
  }
  __syncthreads();
#pragma unroll
  for (int i = 0; i < 2; ++i) {
    int idx = tid + 256 * i;
    int r = idx >> 3, c8 = (idx & 7) * 8;  // r = n-local, c8 = k-local
    size_t o = (size_t)(n0 + r) * kE + k0 + c8;
    *(int4*)&WtHi[o] = *(const int4*)&sHi[r][c8];
    *(int4*)&WtLo[o] = *(const int4*)&sLo[r][c8];
  }
}

// ---------------------------------------------------------------------------
// Transpose V hi/lo planes [b*S+j][h*64+d] -> Vt planes [(b*H+h)*64+d][S].
// One-time; makes attention V staging a pure contiguous copy.
// ---------------------------------------------------------------------------
__global__ __launch_bounds__(256) void transpose_vt(const u16* __restrict__ Ph,
                                                    const u16* __restrict__ Pl,
                                                    u16* __restrict__ Th,
                                                    u16* __restrict__ Tl) {
  __shared__ __align__(16) u16 sh[64][64];
  __shared__ __align__(16) u16 sl[64][64];
  const int tid = threadIdx.x;
  const int j0 = blockIdx.x * 64;
  const int bh = blockIdx.y;
  const int b = bh >> 4, h = bh & 15;
#pragma unroll
  for (int i = 0; i < 2; ++i) {
    int id = tid + 256 * i;
    int r = id >> 3, c8 = (id & 7) * 8;  // r = j-local, c8 = d-local
    size_t src = (size_t)(b * kS + j0 + r) * kE + h * kD + c8;
    union { int4 q; u16 u[8]; } th, tl;
    th.q = *(const int4*)(Ph + src);
    tl.q = *(const int4*)(Pl + src);
#pragma unroll
    for (int e = 0; e < 8; ++e) {
      sh[c8 + e][r] = th.u[e];
      sl[c8 + e][r] = tl.u[e];
    }
  }
  __syncthreads();
#pragma unroll
  for (int i = 0; i < 2; ++i) {
    int id = tid + 256 * i;
    int r = id >> 3, c8 = (id & 7) * 8;  // r = d-local, c8 = j-local
    size_t dst = (size_t)(bh * kD + r) * kS + j0 + c8;
    *(int4*)(Th + dst) = *(const int4*)&sh[r][c8];
    *(int4*)(Tl + dst) = *(const int4*)&sl[r][c8];
  }
}

// ---------------------------------------------------------------------------
// C[M,1024] = A @ W + bias from bf16 hi/lo planes (bf16x3, fp32 accum).
// Double-buffered LDS fed by global_load_async_to_lds_b128.
// ---------------------------------------------------------------------------
template <bool PLANE_OUT>
__global__ __launch_bounds__(256) void gemm_planes(
    const u16* __restrict__ Ahi, const u16* __restrict__ Alo,
    const u16* __restrict__ Bthi, const u16* __restrict__ Btlo,
    const float* __restrict__ bias, u16* __restrict__ Chi, u16* __restrict__ Clo,
    float* __restrict__ Cf) {
  __shared__ __align__(16) u16 sAhi[2][128][32];
  __shared__ __align__(16) u16 sAlo[2][128][32];
  __shared__ __align__(16) u16 sBhi[2][128][32];  // [n][k]
  __shared__ __align__(16) u16 sBlo[2][128][32];

  const int tid = threadIdx.x;
  const int wave = tid >> 5, lane = tid & 31;
  const int wm = wave >> 1, wn = wave & 1;
  const int m0 = blockIdx.y * 128, n0 = blockIdx.x * 128;
  const int rowBase = wm * 32, colBase = wn * 64;

  const unsigned ldsAh = (unsigned)(unsigned long long)&sAhi[0][0][0];
  const unsigned ldsAl = (unsigned)(unsigned long long)&sAlo[0][0][0];
  const unsigned ldsBh = (unsigned)(unsigned long long)&sBhi[0][0][0];
  const unsigned ldsBl = (unsigned)(unsigned long long)&sBlo[0][0][0];

  // Per-thread staging chunks: 2 x 16B per plane per tile.
  const int sr0 = tid >> 2, sc0 = (tid & 3) * 16;
  const int sr1 = (tid + 256) >> 2, sc1 = (tid & 3) * 16;

  auto issue = [&](int buf, int kb) {
    unsigned bufOff = (unsigned)buf * 8192u;
    unsigned a0 = (unsigned)((m0 + sr0) * 2048 + kb * 2 + sc0);
    unsigned a1 = (unsigned)((m0 + sr1) * 2048 + kb * 2 + sc1);
    unsigned b0 = (unsigned)((n0 + sr0) * 2048 + kb * 2 + sc0);
    unsigned b1 = (unsigned)((n0 + sr1) * 2048 + kb * 2 + sc1);
    unsigned l0 = (unsigned)(sr0 * 64 + sc0) + bufOff;
    unsigned l1 = (unsigned)(sr1 * 64 + sc1) + bufOff;
    async16(ldsAh + l0, Ahi, a0);
    async16(ldsAh + l1, Ahi, a1);
    async16(ldsAl + l0, Alo, a0);
    async16(ldsAl + l1, Alo, a1);
    async16(ldsBh + l0, Bthi, b0);
    async16(ldsBh + l1, Bthi, b1);
    async16(ldsBl + l0, Btlo, b0);
    async16(ldsBl + l1, Btlo, b1);
  };

  v8f acc[2][4];
#pragma unroll
  for (int i = 0; i < 2; ++i)
#pragma unroll
    for (int j = 0; j < 4; ++j) acc[i][j] = vzero();

  const int r16 = lane & 15;
  const int k0 = (lane & 16) ? 8 : 0;   // A-fragment K sub-offset
  const int kg = (lane & 16) ? 16 : 0;  // unused; B fragments also use k0

  issue(0, 0);
  int buf = 0;
  for (int kb = 0; kb < kE; kb += 32) {
    wait_async0();
    __syncthreads();
    if (kb + 32 < kE) issue(buf ^ 1, kb + 32);

    Frag ah[2], al[2];
#pragma unroll
    for (int mt = 0; mt < 2; ++mt) {
      int row = rowBase + mt * 16 + r16;
      ah[mt].i4[0] = *(const int4*)&sAhi[buf][row][k0];
      ah[mt].i4[1] = *(const int4*)&sAhi[buf][row][16 + k0];
      al[mt].i4[0] = *(const int4*)&sAlo[buf][row][k0];
      al[mt].i4[1] = *(const int4*)&sAlo[buf][row][16 + k0];
    }
#pragma unroll
    for (int nt = 0; nt < 4; ++nt) {
      int nrow = colBase + nt * 16 + r16;
      Frag bh, bl;
      bh.i4[0] = *(const int4*)&sBhi[buf][nrow][k0];
      bh.i4[1] = *(const int4*)&sBhi[buf][nrow][16 + k0];
      bl.i4[0] = *(const int4*)&sBlo[buf][nrow][k0];
      bl.i4[1] = *(const int4*)&sBlo[buf][nrow][16 + k0];
#pragma unroll
      for (int mt = 0; mt < 2; ++mt) {
        acc[mt][nt] = wmma_bf16(ah[mt].v, bh.v, acc[mt][nt]);
        acc[mt][nt] = wmma_bf16(ah[mt].v, bl.v, acc[mt][nt]);
        acc[mt][nt] = wmma_bf16(al[mt].v, bh.v, acc[mt][nt]);
      }
    }
    buf ^= 1;
  }

  const int rb = (lane >> 4) * 8;
#pragma unroll
  for (int mt = 0; mt < 2; ++mt)
#pragma unroll
    for (int nt = 0; nt < 4; ++nt) {
      int gm = m0 + rowBase + mt * 16 + rb;
      int gn = n0 + colBase + nt * 16 + r16;
      float bv = bias[gn];
#pragma unroll
      for (int v = 0; v < 8; ++v) {
        float val = acc[mt][nt][v] + bv;
        size_t idx = (size_t)(gm + v) * kE + gn;
        if constexpr (PLANE_OUT) {
          u16 h, l;
          split1(val, h, l);
          Chi[idx] = h;
          Clo[idx] = l;
        } else {
          Cf[idx] = val;
        }
      }
    }
}

// ---------------------------------------------------------------------------
// Sliding-window attention from bf16 hi/lo planes. Block = (b, h, 64 q);
// 4 waves x 16 queries, 32-key chunks, online softmax. V chunks stream in
// via async-to-LDS from the pre-transposed Vt planes, overlapping QK+softmax.
// ---------------------------------------------------------------------------
__global__ __launch_bounds__(128) void attn_sliding(
    const u16* __restrict__ Qhi, const u16* __restrict__ Qlo,
    const u16* __restrict__ Khi, const u16* __restrict__ Klo,
    const u16* __restrict__ Vth, const u16* __restrict__ Vtl,
    u16* __restrict__ CxHi, u16* __restrict__ CxLo) {
  __shared__ __align__(16) u16 sVhi[4][64][32];  // [wave][d][j]
  __shared__ __align__(16) u16 sVlo[4][64][32];
  __shared__ __align__(16) float sP[4][16][32];  // [wave][q][j]

  const int tid = threadIdx.x;
  const int wave = tid >> 5, lane = tid & 31;
  const int bid = blockIdx.x;
  const int it = bid & 31;        // S/64 = 32 query tiles
  const int h = (bid >> 5) & 15;  // head
  const int b = bid >> 9;         // batch
  const int q0 = it * 64 + wave * 16;
  const size_t rowOff = (size_t)b * kS;
  const int colOff = h * kD;

  const int r16 = lane & 15;
  const int k0 = (lane & 16) ? 8 : 0;
  const int kg = (lane & 16) ? 16 : 0;
  const int half8 = (lane >> 4) * 8;

  const u16* vth = Vth + (size_t)(b * kH + h) * (kD * kS);
  const u16* vtl = Vtl + (size_t)(b * kH + h) * (kD * kS);
  const unsigned ldsVh = (unsigned)(unsigned long long)&sVhi[wave][0][0];
  const unsigned ldsVl = (unsigned)(unsigned long long)&sVlo[wave][0][0];
  // Per-lane V staging chunk (reused every iteration): 8 x (16B x 2 planes).
  const int vd = lane >> 2;        // d rows handled: vd, vd+8, ... (id>>2 form)
  const int vc = lane & 3;         // 16B chunk within 32-j row

  // Q fragments straight from planes (no conversion).
  Frag qhf[2], qlf[2];
  {
    const u16* qh = Qhi + (rowOff + q0 + r16) * kE + colOff;
    const u16* ql = Qlo + (rowOff + q0 + r16) * kE + colOff;
#pragma unroll
    for (int dI = 0; dI < 2; ++dI) {
      int dB = dI * 32;
      qhf[dI].i4[0] = *(const int4*)(qh + dB + k0);
      qhf[dI].i4[1] = *(const int4*)(qh + dB + 16 + k0);
      qlf[dI].i4[0] = *(const int4*)(ql + dB + k0);
      qlf[dI].i4[1] = *(const int4*)(ql + dB + 16 + k0);
    }
  }

  const float NEG_INF = -__builtin_inff();
  float ms[8], ls[8];
#pragma unroll
  for (int v = 0; v < 8; ++v) {
    ms[v] = NEG_INF;
    ls[v] = 0.f;
  }
  v8f o[4];
#pragma unroll
  for (int i = 0; i < 4; ++i) o[i] = vzero();

  int jstart = q0 - kHalf;
  if (jstart < 0) jstart = 0;
  jstart &= ~31;
  int jend = q0 + 16 + kHalf;
  if (jend > kS) jend = kS;

  for (int jc = jstart; jc < jend; jc += 32) {
    // WAR guard: previous chunk's LDS fragment reads must finish before the
    // async engine overwrites this wave's V region (async != DS ordering).
    wait_ds0();
    // Issue async V staging: 8 d-rows per lane, 16B per row per plane.
    // Runs are 8-j aligned; kS % 8 == 0, so clamped runs are fully masked.
#pragma unroll
    for (int i = 0; i < 8; ++i) {
      int d = vd + i * 8;
      int jr = jc + vc * 8;
      if (jr > kS - 8) jr = kS - 8;
      unsigned go = (unsigned)((d * kS + jr) * 2);
      unsigned lo_ = (unsigned)(d * 64 + vc * 16);
      async16(ldsVh + lo_, vth, go);
      async16(ldsVl + lo_, vtl, go);
    }

    // Scores: S[16q x 32j] as two 16x16 tiles, contraction over D=64.
    v8f s[2];
    s[0] = vzero();
    s[1] = vzero();
#pragma unroll
    for (int jt = 0; jt < 2; ++jt) {
      int j = jc + jt * 16 + r16;
      int jcl = (j < 0) ? 0 : ((j > kS - 1) ? kS - 1 : j);
      size_t off = (rowOff + jcl) * kE + colOff + k0;
#pragma unroll
      for (int dI = 0; dI < 2; ++dI) {
        Frag khf, klf;
        const u16* kh = Khi + off + dI * 32;
        const u16* kl = Klo + off + dI * 32;
        khf.i4[0] = *(const int4*)(kh);
        khf.i4[1] = *(const int4*)(kh + 16);
        klf.i4[0] = *(const int4*)(kl);
        klf.i4[1] = *(const int4*)(kl + 16);
        s[jt] = wmma_bf16(qhf[dI].v, khf.v, s[jt]);
        s[jt] = wmma_bf16(qhf[dI].v, klf.v, s[jt]);
        s[jt] = wmma_bf16(qlf[dI].v, khf.v, s[jt]);
      }
    }

    // Mask + scale + online softmax update (16-lane shuffle row reductions).
#pragma unroll
    for (int v = 0; v < 8; ++v) {
      int q = q0 + half8 + v;
      int j0 = jc + r16, j1 = jc + 16 + r16;
      bool v0 = (j0 >= q - kHalf) && (j0 < q + kHalf) && (j0 >= 0) && (j0 < kS);
      bool v1 = (j1 >= q - kHalf) && (j1 < q + kHalf) && (j1 >= 0) && (j1 < kS);
      float a0 = v0 ? s[0][v] * 0.125f : NEG_INF;
      float a1 = v1 ? s[1][v] * 0.125f : NEG_INF;
      float t = fmaxf(a0, a1);
#pragma unroll
      for (int off = 1; off < 16; off <<= 1) t = fmaxf(t, __shfl_xor(t, off, 32));
      float mn = fmaxf(ms[v], t);
      bool noinf = (mn != NEG_INF);
      float scl = noinf ? __expf(ms[v] - mn) : 1.f;
      float p0 = noinf ? __expf(a0 - mn) : 0.f;
      float p1 = noinf ? __expf(a1 - mn) : 0.f;
      ms[v] = mn;
      float ps = p0 + p1;
#pragma unroll
      for (int off = 1; off < 16; off <<= 1) ps += __shfl_xor(ps, off, 32);
      ls[v] = ls[v] * scl + ps;
#pragma unroll
      for (int d4 = 0; d4 < 4; ++d4) o[d4][v] *= scl;
      sP[wave][half8 + v][r16] = p0;
      sP[wave][half8 + v][16 + r16] = p1;
    }

    // Rebuild P as a 16x32 A-fragment (bf16 hi/lo) from LDS.
    v16bf phi, plo;
    {
      const float* pr = &sP[wave][r16][0];
      float4 fa = *(const float4*)(pr + k0);
      float4 fb = *(const float4*)(pr + k0 + 4);
      float4 fc = *(const float4*)(pr + 16 + k0);
      float4 fd = *(const float4*)(pr + 16 + k0 + 4);
      float t[16];
      t[0] = fa.x;  t[1] = fa.y;  t[2] = fa.z;  t[3] = fa.w;
      t[4] = fb.x;  t[5] = fb.y;  t[6] = fb.z;  t[7] = fb.w;
      t[8] = fc.x;  t[9] = fc.y;  t[10] = fc.z; t[11] = fc.w;
      t[12] = fd.x; t[13] = fd.y; t[14] = fd.z; t[15] = fd.w;
      split16(t, phi, plo);
    }

    // RAW: V chunk must have landed in LDS.
    wait_async0();

    // O += P @ V.
#pragma unroll
    for (int d4 = 0; d4 < 4; ++d4) {
      Frag vh, vl;
      vh.i4[0] = *(const int4*)&sVhi[wave][d4 * 16 + r16][k0];
      vh.i4[1] = *(const int4*)&sVhi[wave][d4 * 16 + r16][16 + k0];
      vl.i4[0] = *(const int4*)&sVlo[wave][d4 * 16 + r16][k0];
      vl.i4[1] = *(const int4*)&sVlo[wave][d4 * 16 + r16][16 + k0];
      o[d4] = wmma_bf16(phi, vh.v, o[d4]);
      o[d4] = wmma_bf16(phi, vl.v, o[d4]);
      o[d4] = wmma_bf16(plo, vh.v, o[d4]);
    }
  }

  // Normalize and write ctx hi/lo planes in [b*S + s][h*64 + d] layout.
  float inv[8];
#pragma unroll
  for (int v = 0; v < 8; ++v) inv[v] = 1.f / ls[v];
#pragma unroll
  for (int d4 = 0; d4 < 4; ++d4)
#pragma unroll
    for (int v = 0; v < 8; ++v) {
      float val = o[d4][v] * inv[v];
      size_t idx = (rowOff + q0 + half8 + v) * kE + colOff + d4 * 16 + r16;
      u16 hh, ll;
      split1(val, hh, ll);
      CxHi[idx] = hh;
      CxLo[idx] = ll;
    }
}

extern "C" void kernel_launch(void* const* d_in, const int* in_sizes, int n_in,
                              void* d_out, int out_size, void* d_ws, size_t ws_size,
                              hipStream_t stream) {
  (void)in_sizes; (void)n_in; (void)out_size; (void)ws_size;
  const float* x  = (const float*)d_in[0];
  const float* Wq = (const float*)d_in[1];
  const float* bq = (const float*)d_in[2];
  const float* Wk = (const float*)d_in[3];
  const float* bk = (const float*)d_in[4];
  const float* Wv = (const float*)d_in[5];
  const float* bv = (const float*)d_in[6];
  const float* Wo = (const float*)d_in[7];
  const float* bo = (const float*)d_in[8];
  float* out = (float*)d_out;

  const size_t P = (size_t)kM * kE;   // activation plane elems (4096*1024)
  const size_t PW = (size_t)kE * kE;  // weight plane elems
  u16* w = (u16*)d_ws;
  u16* xHi = w;            u16* xLo = xHi + P;
  u16* WtQh = xLo + P;     u16* WtQl = WtQh + PW;
  u16* WtKh = WtQl + PW;   u16* WtKl = WtKh + PW;
  u16* WtVh = WtKl + PW;   u16* WtVl = WtVh + PW;
  u16* WtOh = WtVl + PW;   u16* WtOl = WtOh + PW;
  u16* Qh = WtOl + PW;     u16* Ql = Qh + P;
  u16* Kh = Ql + P;        u16* Kl = Kh + P;
  u16* Vh = Kl + P;        u16* Vl = Vh + P;
  u16* CxH = Vl + P;       u16* CxL = CxH + P;
  u16* VtH = CxL + P;      u16* VtL = VtH + P;

  // One-time conversions: x -> planes, W -> transposed planes.
  split_planes<<<(int)(P / 4 / 256), 256, 0, stream>>>(x, xHi, xLo, (int)(P / 4));
  dim3 gT(kE / 64, kE / 64);
  split_transpose<<<gT, 256, 0, stream>>>(Wq, WtQh, WtQl);
  split_transpose<<<gT, 256, 0, stream>>>(Wk, WtKh, WtKl);
  split_transpose<<<gT, 256, 0, stream>>>(Wv, WtVh, WtVl);
  split_transpose<<<gT, 256, 0, stream>>>(Wo, WtOh, WtOl);

  dim3 gGemm(kE / 128, kM / 128);  // (8, 32)
  gemm_planes<true><<<gGemm, 256, 0, stream>>>(xHi, xLo, WtQh, WtQl, bq, Qh, Ql, nullptr);
  gemm_planes<true><<<gGemm, 256, 0, stream>>>(xHi, xLo, WtKh, WtKl, bk, Kh, Kl, nullptr);
  gemm_planes<true><<<gGemm, 256, 0, stream>>>(xHi, xLo, WtVh, WtVl, bv, Vh, Vl, nullptr);
  transpose_vt<<<dim3(kS / 64, kB * kH), 256, 0, stream>>>(Vh, Vl, VtH, VtL);
  attn_sliding<<<kB * kH * (kS / 64), 128, 0, stream>>>(Qh, Ql, Kh, Kl, VtH, VtL,
                                                        CxH, CxL);
  gemm_planes<false><<<gGemm, 256, 0, stream>>>(CxH, CxL, WtOh, WtOl, bo, nullptr,
                                                nullptr, out);
}
